// GNAN_44796508897984
// MI455X (gfx1250) — matrix operand order repaired and checked by reading, not compiled, not run on hardware
//
#include <hip/hip_runtime.h>
#include <hip/hip_bf16.h>

// N=2048, F=64, H=32, O=32, all f32 inputs/outputs.
// Heavy math routed through v_wmma_f32_16x16x32_f16; layer-1 in packed f16.
// ws layout: [ GT = f_sums^T : 32*2048 f16 (128KB) ][ Part : msplit*2048*32 f32 ]

typedef __attribute__((ext_vector_type(16))) _Float16 v16h;
typedef __attribute__((ext_vector_type(8)))  float    v8f;
typedef __attribute__((ext_vector_type(8)))  unsigned u32x8;

#define NN 2048
#define FF 64
#define HH 32
#define OO 32
#define NB (NN / 16)   // 128 row blocks

// single-instruction relu: v_med3_f32(x, 0, FLT_MAX) -- no canonicalize op
static __device__ __forceinline__ float relu_f(float v) {
  return __builtin_amdgcn_fmed3f(v, 0.0f, 3.402823466e38f);
}

static __device__ __forceinline__ v8f wmma_f16(v16h a, v16h b, v8f c) {
  // D = A(16x32 f16) x B(32x16 f16) + C(16x16 f32)
  return __builtin_amdgcn_wmma_f32_16x16x32_f16(false, a, false, b, (short)0, c, false, false);
}

// lane <-> lane^16 value swap (crosses SIMD halves); old=0 (EXEC all ones)
static __device__ __forceinline__ float xor16_swap(float v) {
#if __has_builtin(__builtin_amdgcn_permlanex16)
  int r = __builtin_amdgcn_permlanex16(0, __builtin_bit_cast(int, v),
                                       0x76543210, 0xfedcba98, false, false);
  return __builtin_bit_cast(float, r);
#else
  return __shfl_xor(v, 16, 32);
#endif
}

// splat one f16 into all 16 lanes of a v16h via integer replication
static __device__ __forceinline__ v16h splat_h(_Float16 h) {
  const unsigned d = (unsigned)__builtin_bit_cast(unsigned short, h) * 0x00010001u;
  const u32x8 v = {d, d, d, d, d, d, d, d};
  return __builtin_bit_cast(v16h, v);
}

// ---------------------------------------------------------------------------
// Kernel 1: f_sums; one wave per 16-node block. Transposed WMMA chain:
// H2^T = relu(W2f^T @ H1^T + b2), acc^T += W3f^T @ H2^T (C-accumulated over f).
// Output written transposed + f16: GT[o][n].
// ---------------------------------------------------------------------------
__global__ __launch_bounds__(32) void fsums_kernel(
    const float* __restrict__ x,
    const float* __restrict__ fW1, const float* __restrict__ fb1,
    const float* __restrict__ fW2, const float* __restrict__ fb2,
    const float* __restrict__ fW3, const float* __restrict__ fb3,
    _Float16* __restrict__ GT)
{
  const int lane = threadIdx.x & 31;
  const int lo = lane & 15;      // node index within block / matrix column
  const int hi = lane >> 4;      // K-half selector
  const int n0 = blockIdx.x * 16;

  v8f acc0 = {};
  v8f acc1 = {};

  for (int f = 0; f < FF; ++f) {
    const float xv = x[(n0 + lo) * FF + f];
    const float* w1 = fW1 + f * HH;   // fW1[f,0,:]
    const float* b1 = fb1 + f * HH;

    // B = H1^T (32k x 16 nodes): lane holds its node, K = 16*hi + j
    v16h bh;
#pragma unroll
    for (int j = 0; j < 16; ++j) {
      const int k = 16 * hi + j;
      bh[j] = (_Float16)relu_f(fmaf(xv, w1[k], b1[k]));
    }

    // A tiles: W2^T and W3^T rows (A layout: k = 16*hi + j)
    v16h a20, a21, a30, a31;
#pragma unroll
    for (int j = 0; j < 16; ++j) {
      const int kA = 16 * hi + j;
      a20[j] = (_Float16)fW2[(f * HH + kA) * HH + lo];
      a21[j] = (_Float16)fW2[(f * HH + kA) * HH + 16 + lo];
      a30[j] = (_Float16)fW3[(f * HH + kA) * OO + lo];
      a31[j] = (_Float16)fW3[(f * HH + kA) * OO + 16 + lo];
    }

    // C = b2 broadcast (row M=r -> channel g = 16t + 8*hi + r)
    v8f c0, c1;
#pragma unroll
    for (int r = 0; r < 8; ++r) {
      c0[r] = fb2[f * HH + 8 * hi + r];
      c1[r] = fb2[f * HH + 16 + 8 * hi + r];
    }

    v8f d0 = wmma_f16(a20, bh, c0);   // H2^T rows g=0..15
    v8f d1 = wmma_f16(a21, bh, c1);   // H2^T rows g=16..31
#pragma unroll
    for (int r = 0; r < 8; ++r) { d0[r] = relu_f(d0[r]); d1[r] = relu_f(d1[r]); }

    // Repack D layout -> B layout (H2^T as B matrix) with one xor-16 swap
    v16h b3;
#pragma unroll
    for (int r = 0; r < 8; ++r) {
      const float p0 = xor16_swap(d0[r]);
      const float p1 = xor16_swap(d1[r]);
      b3[r]     = (_Float16)(hi ? p1 : d0[r]);   // k = 16*hi + r
      b3[8 + r] = (_Float16)(hi ? d1[r] : p0);   // k = 16*hi + 8 + r
    }

    acc0 = wmma_f16(a30, b3, acc0);   // o = 0..15
    acc1 = wmma_f16(a31, b3, acc1);   // o = 16..31
  }

  // add sum_f fb3[f,o] (zeros in setup, kept for generality); store f16 ^T
#pragma unroll
  for (int r = 0; r < 8; ++r) {
    const int o0 = 8 * hi + r;
    const int o1 = 16 + 8 * hi + r;
    float s0 = 0.0f, s1 = 0.0f;
    for (int f = 0; f < FF; ++f) { s0 += fb3[f * OO + o0]; s1 += fb3[f * OO + o1]; }
    GT[(size_t)o0 * NN + n0 + lo] = (_Float16)(acc0[r] + s0);
    GT[(size_t)o1 * NN + n0 + lo] = (_Float16)(acc1[r] + s1);
  }
}

// ---------------------------------------------------------------------------
// Kernel 2: rho(d)/norm fused with out = srho @ f_sums. One wave per
// (16-row block i, m-chunk s). Partial [16 x 32] tiles to workspace.
// ---------------------------------------------------------------------------
__global__ __launch_bounds__(32) void rho_kernel(
    const float* __restrict__ D,
    const float* __restrict__ Norm,
    const float* __restrict__ rW1, const float* __restrict__ rb1,
    const float* __restrict__ rW2, const float* __restrict__ rb2,
    const float* __restrict__ rW3, const float* __restrict__ rb3,
    const _Float16* __restrict__ GT,
    float* __restrict__ Part,
    int chunk)
{
  const int lane = threadIdx.x & 31;
  const int lo = lane & 15;
  const int hi = lane >> 4;
  const int ib = blockIdx.x;
  const int i0 = ib * 16;
  const int s  = blockIdx.y;
  const int m0 = s * chunk;

  // ---- per-lane constants (loop-invariant) ----
  const v16h zero16 = {};
  v16h w1p, b1p;                       // layer-1 weights packed f16, K=16*hi+j
#pragma unroll
  for (int j = 0; j < 16; ++j) {
    w1p[j] = (_Float16)rW1[16 * hi + j];
    b1p[j] = (_Float16)rb1[16 * hi + j];
  }
  v16h a20, a21;                       // W2^T tiles in A layout
#pragma unroll
  for (int j = 0; j < 16; ++j) {
    const int kA = 16 * hi + j;
    a20[j] = (_Float16)rW2[kA * HH + lo];
    a21[j] = (_Float16)rW2[kA * HH + 16 + lo];
  }
  v8f cb0, cb1;                        // b2 broadcast as C
  float w30[8], w31[8];                // w3 slices matching D rows
#pragma unroll
  for (int r = 0; r < 8; ++r) {
    cb0[r] = rb2[8 * hi + r];
    cb1[r] = rb2[16 + 8 * hi + r];
    w30[r] = rW3[8 * hi + r];
    w31[r] = rW3[16 + 8 * hi + r];
  }
  const float b3s = rb3[0];

  const float*    drow = D    + (size_t)(i0 + lo) * NN;
  const float*    nrow = Norm + (size_t)(i0 + lo) * NN;
  const _Float16* gr0  = GT + (size_t)lo * NN;         // f_sums^T row o=lo
  const _Float16* gr1  = GT + (size_t)(16 + lo) * NN;  // f_sums^T row o=16+lo

  v8f accF0 = {}, accF1 = {};          // out partial [16i x 32o] in D layout

  for (int mb = m0; mb < m0 + chunk; mb += 32) {
    __builtin_prefetch(drow + mb + 32, 0, 1);
    __builtin_prefetch(nrow + mb + 32, 0, 1);

    // B tiles of f_sums for the final GEMM: contiguous f16 vector loads
    const v16h g0 = *(const v16h*)(gr0 + mb + 16 * hi);
    const v16h g1 = *(const v16h*)(gr1 + mb + 16 * hi);

    v16h aS = {};                      // SRho[16i x 32m] tile, A layout
#pragma unroll
    for (int g = 0; g < 32; ++g) {
      const int m = mb + g;
      const float dv = __builtin_nontemporal_load(drow + m);

      // layer 1 in packed f16: B = H1^T, lane's entry i=lo, K = 16*hi + j
      const v16h dsplat = splat_h((_Float16)dv);
      v16h bh = __builtin_elementwise_fma(dsplat, w1p, b1p); // v_pk_fma_f16
      bh = __builtin_elementwise_max(bh, zero16);            // v_pk_max (relu)

      v8f d0 = wmma_f16(a20, bh, cb0);     // H2^T channels 0..15
      v8f d1 = wmma_f16(a21, bh, cb1);     // H2^T channels 16..31

      // layer 3: rho = relu(H2) . w3  (lane-pair holds all 32 channels)
      float p = 0.0f;
#pragma unroll
      for (int r = 0; r < 8; ++r) {
        p = fmaf(relu_f(d0[r]), w30[r], p);
        p = fmaf(relu_f(d1[r]), w31[r], p);
      }
      p += xor16_swap(p);

      const float nv = __builtin_nontemporal_load(nrow + m);
      const float sr = (p + b3s) * __builtin_amdgcn_rcpf(nv);
      const _Float16 hs = (_Float16)sr;

      // deposit into SRho A-tile: lane half 'ah' owns K=g at slot j
      const int ah = g >> 4;
      const int j  = g & 15;
      aS[j] = (hi == ah) ? hs : aS[j];
    }

    // out[16i,32o] += SRho[16i,32m] @ G[32m,32o]
    accF0 = wmma_f16(aS, g0, accF0);
    accF1 = wmma_f16(aS, g1, accF1);
  }

  // store partial tile: D layout -> Part[s][ib][i_local][o]
  float* pp = Part + (size_t)(s * NB + ib) * 16 * OO;
#pragma unroll
  for (int r = 0; r < 8; ++r) {
    const int irow = 8 * hi + r;
    pp[irow * OO + lo]      = accF0[r];
    pp[irow * OO + 16 + lo] = accF1[r];
  }
}

// ---------------------------------------------------------------------------
// Kernel 3: deterministic reduction of m-chunk partials.
// ---------------------------------------------------------------------------
__global__ void reduce_kernel(const float* __restrict__ Part,
                              float* __restrict__ out, int msplit)
{
  const int t = blockIdx.x * blockDim.x + threadIdx.x;
  if (t >= NN * OO) return;
  const int i = t / OO, o = t % OO;
  const int ib = i >> 4, r = i & 15;
  float sum = 0.0f;
  for (int c = 0; c < msplit; ++c)
    sum += Part[((size_t)(c * NB + ib) * 16 + r) * OO + o];
  out[t] = sum;
}

extern "C" void kernel_launch(void* const* d_in, const int* in_sizes, int n_in,
                              void* d_out, int out_size, void* d_ws, size_t ws_size,
                              hipStream_t stream)
{
  const float* x    = (const float*)d_in[0];
  const float* dmat = (const float*)d_in[1];
  const float* nmat = (const float*)d_in[2];
  const float* fW1  = (const float*)d_in[3];
  const float* fb1  = (const float*)d_in[4];
  const float* fW2  = (const float*)d_in[5];
  const float* fb2  = (const float*)d_in[6];
  const float* fW3  = (const float*)d_in[7];
  const float* fb3  = (const float*)d_in[8];
  const float* rW1  = (const float*)d_in[9];
  const float* rb1  = (const float*)d_in[10];
  const float* rW2  = (const float*)d_in[11];
  const float* rb2  = (const float*)d_in[12];
  const float* rW3  = (const float*)d_in[13];
  const float* rb3  = (const float*)d_in[14];

  // ws: GT (f16, 128KB) then Part (f32). Shrink m-split if ws is tight.
  const size_t gt_bytes = (size_t)NN * OO * sizeof(_Float16);
  int msplit = 32;                 // 4096 waves; chunk=64 (multiple of 32)
  while (msplit > 1 &&
         gt_bytes + (size_t)msplit * NN * OO * sizeof(float) > ws_size)
    msplit >>= 1;
  const int chunk = NN / msplit;

  _Float16* GT  = (_Float16*)d_ws;
  float*    Part = (float*)((char*)d_ws + gt_bytes);

  fsums_kernel<<<dim3(NB), dim3(32), 0, stream>>>(x, fW1, fb1, fW2, fb2, fW3, fb3, GT);
  rho_kernel<<<dim3(NB, msplit), dim3(32), 0, stream>>>(dmat, nmat, rW1, rb1, rW2, rb2,
                                                        rW3, rb3, GT, Part, chunk);
  reduce_kernel<<<dim3((NN * OO + 255) / 256), dim3(256), 0, stream>>>(Part, (float*)d_out, msplit);
}
